// MoEGate_63754494542474
// MI455X (gfx1250) — compile-verified, matrix operand order checked by reading, not run on hardware
//
#include <hip/hip_runtime.h>
#include <hip/hip_bf16.h>

typedef __attribute__((ext_vector_type(16))) __bf16 v16bf;
typedef __attribute__((ext_vector_type(8)))  __bf16 v8bf;
typedef __attribute__((ext_vector_type(4)))  __bf16 v4bf;
typedef __attribute__((ext_vector_type(8)))  float  v8f;
typedef __attribute__((ext_vector_type(4)))  float  v4f;

#define D_DIM        2048
#define E_DIM        64
#define TOPK_N       8
#define KCHUNK       512
#define LDS_WSTRIDE  520   // bf16 elems per expert row in LDS (8-elem pad: 1040B = 65*16)
#define BLOCK_TOKENS 64
#define THREADS      128
#define SCORE_STRIDE 68    // padded float row for per-token scores (272B = 17*16, b128-aligned)

// ---------------------------------------------------------------------------
// Kernel 1: convert router weight [E, D] fp32 -> bf16 (once per launch, 512KB->256KB)
// ---------------------------------------------------------------------------
__global__ void wcvt_kernel(const float* __restrict__ w, __bf16* __restrict__ wbf, int n4) {
    int i = blockIdx.x * blockDim.x + threadIdx.x;
    if (i < n4) {
        const v4f f = ((const v4f*)w)[i];
        v4bf o;
        o[0] = (__bf16)f.x; o[1] = (__bf16)f.y; o[2] = (__bf16)f.z; o[3] = (__bf16)f.w;
        ((v4bf*)wbf)[i] = o;
    }
}

// ---------------------------------------------------------------------------
// Kernel 2: fused gate GEMM (WMMA bf16) + top-8 softmax + renorm
// grid = T/64 blocks, 128 threads (4 wave32); each wave: 16 tokens x 64 experts
// ---------------------------------------------------------------------------
__global__ __launch_bounds__(THREADS)
void moe_gate_kernel(const float* __restrict__ x, const __bf16* __restrict__ wbf,
                     float* __restrict__ out_w, float* __restrict__ out_i) {
    __shared__ __align__(16) char smem[E_DIM * LDS_WSTRIDE * 2];  // 66560 B
    __bf16* lds_w = (__bf16*)smem;                                 // staged W chunk
    float*  lds_s = (float*)smem;                                  // reused for scores

    const int tid  = threadIdx.x;
    const int lane = tid & 31;
    const int wv   = tid >> 5;        // wave 0..3
    const int l16  = lane & 15;
    const int hi   = lane >> 4;       // 0 | 1 (half-wave)
    const int koff = hi * 8;          // ISA 16-bit A/B fragment K offset per half-wave

    const int blockTok = blockIdx.x * BLOCK_TOKENS;
    const int tokRow   = blockTok + wv * 16 + l16;
    const float* xrow  = x + (size_t)tokRow * D_DIM + koff;

    v8f acc[4];
    #pragma unroll
    for (int nt = 0; nt < 4; ++nt) acc[nt] = (v8f){0.f,0.f,0.f,0.f,0.f,0.f,0.f,0.f};

    const int GROUPS_PER_ROW = KCHUNK / 8;  // 64 x uint4 groups per expert row

    for (int kc = 0; kc < D_DIM; kc += KCHUNK) {
        __syncthreads();
        // Cooperative stage: W chunk [64 x 512] bf16 -> LDS, rows padded to LDS_WSTRIDE
        for (int i = tid; i < E_DIM * GROUPS_PER_ROW; i += THREADS) {
            const int row = i / GROUPS_PER_ROW;
            const int g   = i % GROUPS_PER_ROW;
            *(uint4*)(lds_w + row * LDS_WSTRIDE + g * 8) =
                *(const uint4*)(wbf + (size_t)row * D_DIM + kc + g * 8);
        }
        __syncthreads();

        for (int k = 0; k < KCHUNK; k += 32) {
            // ---- A fragment: 16 tokens x 32 K; streamed fp32 (NT) -> bf16 packs ----
            const float* ap = xrow + kc + k;
            const v4f f0 = __builtin_nontemporal_load((const v4f*)(ap));
            const v4f f1 = __builtin_nontemporal_load((const v4f*)(ap + 4));
            const v4f f2 = __builtin_nontemporal_load((const v4f*)(ap + 16));
            const v4f f3 = __builtin_nontemporal_load((const v4f*)(ap + 20));
            v16bf a;
            a[0]  = (__bf16)f0.x; a[1]  = (__bf16)f0.y; a[2]  = (__bf16)f0.z; a[3]  = (__bf16)f0.w;
            a[4]  = (__bf16)f1.x; a[5]  = (__bf16)f1.y; a[6]  = (__bf16)f1.z; a[7]  = (__bf16)f1.w;
            a[8]  = (__bf16)f2.x; a[9]  = (__bf16)f2.y; a[10] = (__bf16)f2.z; a[11] = (__bf16)f2.w;
            a[12] = (__bf16)f3.x; a[13] = (__bf16)f3.y; a[14] = (__bf16)f3.z; a[15] = (__bf16)f3.w;

            // ---- 4 expert tiles of 16: B fragment from LDS + WMMA ----
            #pragma unroll
            for (int nt = 0; nt < 4; ++nt) {
                const __bf16* wr = lds_w + (nt * 16 + l16) * LDS_WSTRIDE + k + koff;
                const v8bf blo = *(const v8bf*)(wr);
                const v8bf bhi = *(const v8bf*)(wr + 16);
                const v16bf b = __builtin_shufflevector(blo, bhi,
                        0,1,2,3,4,5,6,7,8,9,10,11,12,13,14,15);
                acc[nt] = __builtin_amdgcn_wmma_f32_16x16x32_bf16(
                        /*neg_a=*/false, a, /*neg_b=*/false, b,
                        /*c_mod=*/(short)0, acc[nt],
                        /*reuse_a=*/false, /*reuse_b=*/false);
            }
        }
    }

    // ---- spill logits to LDS: token-major rows of 64 (padded) ----
    __syncthreads();
    #pragma unroll
    for (int nt = 0; nt < 4; ++nt) {
        #pragma unroll
        for (int r = 0; r < 8; ++r) {
            const int t = wv * 16 + r + hi * 8;           // C-layout: lane half selects M+8
            lds_s[t * SCORE_STRIDE + nt * 16 + l16] = acc[nt][r];
        }
    }
    __syncthreads();

    // ---- per-token top-8 + softmax-over-top8 (one thread per token) ----
    // softmax Z cancels under NORM_TOPK (epsilon 1e-20 is < fp32 eps relative),
    // and exp is monotonic -> select top-8 directly on logits, exp only 8 values.
    if (tid < BLOCK_TOKENS) {
        float* row = lds_s + tid * SCORE_STRIDE;
        const float NEG_INF = -__builtin_inff();

        float w8[TOPK_N];
        int   i8[TOPK_N];
        #pragma unroll
        for (int j = 0; j < TOPK_N; ++j) {
            float best = NEG_INF; int bi = 0;
            for (int q = 0; q < E_DIM / 4; ++q) {          // vectorized masked argmax
                const v4f v = *(const v4f*)(row + q * 4);
                if (v.x > best) { best = v.x; bi = q * 4 + 0; }
                if (v.y > best) { best = v.y; bi = q * 4 + 1; }
                if (v.z > best) { best = v.z; bi = q * 4 + 2; }
                if (v.w > best) { best = v.w; bi = q * 4 + 3; }
            }
            row[bi] = NEG_INF;                             // mask for next pass
            w8[j] = best;
            i8[j] = bi;
        }

        const float m = w8[0];                             // global max = first pick
        float s8 = 0.f;
        #pragma unroll
        for (int j = 0; j < TOPK_N; ++j) { w8[j] = __expf(w8[j] - m); s8 += w8[j]; }
        const float dn = 1.f / s8;

        const int gtok = blockTok + tid;
        #pragma unroll
        for (int j = 0; j < TOPK_N; ++j) {
            out_w[(size_t)gtok * TOPK_N + j] = w8[j] * dn;
            out_i[(size_t)gtok * TOPK_N + j] = (float)i8[j];
        }
    }
}

extern "C" void kernel_launch(void* const* d_in, const int* in_sizes, int n_in,
                              void* d_out, int out_size, void* d_ws, size_t ws_size,
                              hipStream_t stream) {
    const float* x = (const float*)d_in[0];   // [B,S,D] fp32
    const float* w = (const float*)d_in[1];   // [E,D]   fp32
    const int T = in_sizes[0] / D_DIM;        // 16384 tokens

    __bf16* wbf   = (__bf16*)d_ws;            // E*D bf16 = 256 KB scratch
    float*  out_w = (float*)d_out;            // [T,8] weights
    float*  out_i = out_w + (size_t)T * TOPK_N; // [T,8] indices (as float values)

    const int n4 = (E_DIM * D_DIM) / 4;
    wcvt_kernel<<<(n4 + 255) / 256, 256, 0, stream>>>(w, wbf, n4);
    moe_gate_kernel<<<T / BLOCK_TOKENS, THREADS, 0, stream>>>(x, wbf, out_w, out_i);
}
